// BlockAttnResTransformerBlock_14250701488713
// MI455X (gfx1250) — matrix operand order repaired and checked by reading, not compiled, not run on hardware
//
#include <hip/hip_runtime.h>
#include <hip/hip_bf16.h>
#include <stdint.h>

// ---------- types ----------
typedef __attribute__((ext_vector_type(16))) __bf16         v16bf;
typedef __attribute__((ext_vector_type(8)))  float          v8f;
typedef __attribute__((ext_vector_type(4)))  unsigned int   u32x4;   // POD 16B
typedef __attribute__((ext_vector_type(4)))  unsigned short u16x4;   // POD 8B

#define D_DIM 2048
#define NCOMP 4
#define NBLK  5
#define EPS_F 1e-6f

__device__ __forceinline__ unsigned short f2bf(float f) {
  unsigned u = __float_as_uint(f);
  u += 0x7FFFu + ((u >> 16) & 1u);           // round-to-nearest-even
  return (unsigned short)(u >> 16);
}

// ---------------------------------------------------------------------------
// fp32 -> bf16 weight conversion (tiny: 2 x 16.7MB read, done once per call)
// ---------------------------------------------------------------------------
__global__ void cvt_f32_to_bf16(const float* __restrict__ in,
                                unsigned short* __restrict__ out, int n4) {
  int i = blockIdx.x * blockDim.x + threadIdx.x;
  if (i >= n4) return;
  float4 v = ((const float4*)in)[i];
  u16x4 o;
  o.x = f2bf(v.x); o.y = f2bf(v.y); o.z = f2bf(v.z); o.w = f2bf(v.w);
  ((u16x4*)out)[i] = o;
}

// ---------------------------------------------------------------------------
// Inter-block attention + fused outer rmsnorm, bf16 output (GEMM A-matrix).
// One wave32 per (b,t) row; 4 waves per block; V rows staged in LDS so global
// V is streamed exactly once (bandwidth floor).  Dynamic LDS = 4*5*8KB = 160KB
// (CDNA5 WGP has 320KB).
// ---------------------------------------------------------------------------
__global__ __launch_bounds__(128) void attn_rmsnorm_kernel(
    const float* __restrict__ comp,   // [NCOMP, rows, D]
    const float* __restrict__ part,   // [rows, D]
    const float* __restrict__ q,      // [D]  *_res_query
    const float* __restrict__ wres,   // [D]  *_res_norm_w (K-norm weight)
    const float* __restrict__ wout,   // [D]  outer norm weight
    unsigned short* __restrict__ xn,  // [rows, D] bf16 out
    int rows) {
  extern __shared__ float vlds_all[];
  const int lane = threadIdx.x & 31;
  const int wave = threadIdx.x >> 5;
  const int row  = blockIdx.x * 4 + wave;
  float* vlds = vlds_all + (size_t)wave * NBLK * D_DIM;

  // ---- pass 1: stream V -> LDS, accumulate ss and q·(w*x) per block n ----
  float logits[NBLK];
#pragma unroll
  for (int n = 0; n < NBLK; ++n) {
    const float* src = (n < NCOMP) ? comp + ((size_t)n * rows + row) * D_DIM
                                   : part + (size_t)row * D_DIM;
    float ss = 0.f, dot = 0.f;
#pragma unroll
    for (int c = lane * 4; c < D_DIM; c += 128) {
      float4 v  = *(const float4*)(src + c);
      float4 qq = *(const float4*)(q + c);
      float4 ww = *(const float4*)(wres + c);
      *(float4*)(vlds + n * D_DIM + c) = v;
      ss  += v.x * v.x + v.y * v.y + v.z * v.z + v.w * v.w;
      dot += qq.x * ww.x * v.x + qq.y * ww.y * v.y +
             qq.z * ww.z * v.z + qq.w * ww.w * v.w;
    }
#pragma unroll
    for (int m = 16; m >= 1; m >>= 1) {
      ss  += __shfl_xor(ss, m, 32);
      dot += __shfl_xor(dot, m, 32);
    }
    // logit = q·rmsnorm(x,w) = (Σ q*w*x) * rsqrt(mean(x^2)+eps)
    logits[n] = dot * rsqrtf(ss * (1.0f / D_DIM) + EPS_F);
  }

  // ---- softmax over the 5 blocks (replicated in every lane) ----
  float mx = logits[0];
#pragma unroll
  for (int n = 1; n < NBLK; ++n) mx = fmaxf(mx, logits[n]);
  float alpha[NBLK];
  float den = 0.f;
#pragma unroll
  for (int n = 0; n < NBLK; ++n) { alpha[n] = expf(logits[n] - mx); den += alpha[n]; }
  const float inv = 1.0f / den;
#pragma unroll
  for (int n = 0; n < NBLK; ++n) alpha[n] *= inv;

  // ---- pass 2: h = Σ αn·Vn from LDS; fused rmsnorm(h)*wout -> bf16 ----
  float4 hreg[D_DIM / 128];
  float hss = 0.f;
#pragma unroll
  for (int i = 0, c = lane * 4; c < D_DIM; c += 128, ++i) {
    float4 acc = {0.f, 0.f, 0.f, 0.f};
#pragma unroll
    for (int n = 0; n < NBLK; ++n) {
      float4 v = *(const float4*)(vlds + n * D_DIM + c);
      acc.x += alpha[n] * v.x; acc.y += alpha[n] * v.y;
      acc.z += alpha[n] * v.z; acc.w += alpha[n] * v.w;
    }
    hreg[i] = acc;
    hss += acc.x * acc.x + acc.y * acc.y + acc.z * acc.z + acc.w * acc.w;
  }
#pragma unroll
  for (int m = 16; m >= 1; m >>= 1) hss += __shfl_xor(hss, m, 32);
  const float rms = rsqrtf(hss * (1.0f / D_DIM) + EPS_F);

  unsigned short* dst = xn + (size_t)row * D_DIM;
#pragma unroll
  for (int i = 0, c = lane * 4; c < D_DIM; c += 128, ++i) {
    float4 hv = hreg[i];
    float4 ww = *(const float4*)(wout + c);
    u16x4 o;
    o.x = f2bf(hv.x * rms * ww.x);
    o.y = f2bf(hv.y * rms * ww.y);
    o.z = f2bf(hv.z * rms * ww.z);
    o.w = f2bf(hv.w * rms * ww.w);
    *(u16x4*)(dst + c) = o;
  }
}

// ---------------------------------------------------------------------------
// C[M,N] = res[M,N] + Xbf16[M,K] @ Wbf16[N,K]^T   (out = res + x @ W.T)
// v_wmma_f32_16x16x32_bf16, 128x128 block tile, BK=32.
// Double-buffered LDS filled by CDNA5 async DMA (global_load_async_to_lds),
// tracked with ASYNCcnt: next K-tile streams into LDS while WMMAs run.
// 8 waves arranged 2(M) x 4(N); each wave: 4x2 fragments (64x32 per wave).
// ---------------------------------------------------------------------------
#define BM 128
#define BN 128
#define BK 32
#define KPAD 8
#define LDT (BK + KPAD)  // 40 bf16 per row -> 80B row stride (16B aligned)

__global__ __launch_bounds__(256) void gemm_bf16_wmma(
    const unsigned short* __restrict__ X,  // [M][K] bf16
    const unsigned short* __restrict__ W,  // [N][K] bf16
    const float* res, float* out, int M, int N, int K) {
  __shared__ unsigned short As[2][BM * LDT];
  __shared__ unsigned short Bs[2][BN * LDT];

  const int tid   = threadIdx.x;
  const int lane  = tid & 31;
  const int wave  = tid >> 5;
  const int warpM = wave & 1;   // 2 waves along M
  const int warpN = wave >> 1;  // 4 waves along N
  const int half  = lane >> 4;  // lane half selects K-group (ISA layout)
  const int ln    = lane & 15;

  const int m0 = blockIdx.y * BM;
  const int n0 = blockIdx.x * BN;

  // async global->LDS staging: each thread owns one 32B run (two b128 ops,
  // offset:0 / offset:16 -- INST_OFFSET applies to BOTH lds and global addr)
  const int srow = tid >> 1;        // 0..127
  const int sseg = (tid & 1) * 16;  // 0 | 16
  const unsigned gaByte = (unsigned)(((size_t)(m0 + srow) * K + sseg) * 2u);
  const unsigned gbByte = (unsigned)(((size_t)(n0 + srow) * K + sseg) * 2u);
  const unsigned ldsA   = (unsigned)(uintptr_t)&As[0][srow * LDT + sseg];
  const unsigned ldsB   = (unsigned)(uintptr_t)&Bs[0][srow * LDT + sseg];
  const unsigned bufStr = (unsigned)(BM * LDT * 2);  // bytes per buffer

  auto issue_tile = [&](int buf, int kb) {
    const unsigned ga = gaByte + (unsigned)(kb * 2);
    const unsigned gb = gbByte + (unsigned)(kb * 2);
    const unsigned la = ldsA + (unsigned)buf * bufStr;
    const unsigned lb = ldsB + (unsigned)buf * bufStr;
    asm volatile("global_load_async_to_lds_b128 %0, %1, %2"
                 :: "v"(la), "v"(ga), "s"(X) : "memory");
    asm volatile("global_load_async_to_lds_b128 %0, %1, %2 offset:16"
                 :: "v"(la), "v"(ga), "s"(X) : "memory");
    asm volatile("global_load_async_to_lds_b128 %0, %1, %2"
                 :: "v"(lb), "v"(gb), "s"(W) : "memory");
    asm volatile("global_load_async_to_lds_b128 %0, %1, %2 offset:16"
                 :: "v"(lb), "v"(gb), "s"(W) : "memory");
  };

  v8f acc[4][2];
#pragma unroll
  for (int tm = 0; tm < 4; ++tm)
#pragma unroll
    for (int tn = 0; tn < 2; ++tn) acc[tm][tn] = 0.f;

  issue_tile(0, 0);
  asm volatile("s_wait_asynccnt 0x0" ::: "memory");
  __syncthreads();
  int cur = 0;

  union Frag { u32x4 u[2]; v16bf v; };

  for (int kb = 0; kb < K; kb += BK) {
    const bool more = (kb + BK) < K;
    if (more) issue_tile(cur ^ 1, kb + BK);   // DMA next tile during compute

    v16bf afr[4], bfr[2];
    // 16-bit A 16x32 layout: lanes 0-15 -> K{0..7,16..23}, lanes 16-31 -> +8
#pragma unroll
    for (int tm = 0; tm < 4; ++tm) {
      const int mrow = warpM * 64 + tm * 16 + ln;
      Frag f;
      f.u[0] = *(const u32x4*)&As[cur][mrow * LDT + 8 * half];
      f.u[1] = *(const u32x4*)&As[cur][mrow * LDT + 16 + 8 * half];
      afr[tm] = f.v;
    }
    // 16-bit B 32x16 layout: lanes 0-15 -> K{0..7,16..23}, lanes 16-31 -> +8
#pragma unroll
    for (int tn = 0; tn < 2; ++tn) {
      const int nrow = warpN * 32 + tn * 16 + ln;
      Frag f;
      f.u[0] = *(const u32x4*)&Bs[cur][nrow * LDT + 8 * half];
      f.u[1] = *(const u32x4*)&Bs[cur][nrow * LDT + 16 + 8 * half];
      bfr[tn] = f.v;
    }
#pragma unroll
    for (int tm = 0; tm < 4; ++tm)
#pragma unroll
      for (int tn = 0; tn < 2; ++tn)
        acc[tm][tn] = __builtin_amdgcn_wmma_f32_16x16x32_bf16(
            false, afr[tm], false, bfr[tn], (short)0, acc[tm][tn], false, false);

    if (more) {
      asm volatile("s_wait_asynccnt 0x0" ::: "memory");  // next tile landed
      __syncthreads();
      cur ^= 1;
    }
  }

  // epilogue: C/D frag VGPR r -> row r (lanes 0-15) / r+8 (lanes 16-31)
#pragma unroll
  for (int tm = 0; tm < 4; ++tm)
#pragma unroll
    for (int tn = 0; tn < 2; ++tn)
#pragma unroll
      for (int r = 0; r < 8; ++r) {
        const int gm = m0 + warpM * 64 + tm * 16 + r + 8 * half;
        const int gn = n0 + warpN * 32 + tn * 16 + ln;
        const size_t idx = (size_t)gm * N + gn;
        out[idx] = res[idx] + acc[tm][tn][r];
      }
}

// ---------------------------------------------------------------------------
extern "C" void kernel_launch(void* const* d_in, const int* in_sizes, int n_in,
                              void* d_out, int out_size, void* d_ws, size_t ws_size,
                              hipStream_t stream) {
  const float* comp        = (const float*)d_in[0];
  const float* part        = (const float*)d_in[1];
  const float* attn_norm_w = (const float*)d_in[2];
  const float* attn_proj   = (const float*)d_in[3];
  const float* mlp_norm_w  = (const float*)d_in[4];
  const float* mlp_proj    = (const float*)d_in[5];
  const float* attn_q      = (const float*)d_in[6];
  const float* attn_res_nw = (const float*)d_in[7];
  const float* mlp_q       = (const float*)d_in[8];
  const float* mlp_res_nw  = (const float*)d_in[9];
  // d_in[10] = layer_in_block (fixed path in reference; unused)

  const int d    = D_DIM;
  const int rows = in_sizes[1] / d;  // b*t = 8192
  float* out = (float*)d_out;

  // workspace: [Wattn bf16][Wmlp bf16][X bf16]
  unsigned short* wbfA = (unsigned short*)d_ws;
  unsigned short* wbfM = wbfA + (size_t)d * d;
  unsigned short* xnB  = wbfM + (size_t)d * d;

  {
    const int n4  = d * d / 4;
    const int thr = 256;
    const int blk = (n4 + thr - 1) / thr;
    cvt_f32_to_bf16<<<blk, thr, 0, stream>>>(attn_proj, wbfA, n4);
    cvt_f32_to_bf16<<<blk, thr, 0, stream>>>(mlp_proj, wbfM, n4);
  }

  const size_t attn_lds = (size_t)4 * NBLK * D_DIM * sizeof(float);  // 160KB
  const dim3 gemm_grid(d / BN, rows / BM);

  // sublayer 1 (attn): V = [completed..., partial]
  attn_rmsnorm_kernel<<<rows / 4, 128, attn_lds, stream>>>(
      comp, part, attn_q, attn_res_nw, attn_norm_w, xnB, rows);
  gemm_bf16_wmma<<<gemm_grid, 256, 0, stream>>>(xnB, wbfA, part, out, rows, d, d);

  // sublayer 2 (mlp): V = [completed..., updated partial]; in-place residual
  attn_rmsnorm_kernel<<<rows / 4, 128, attn_lds, stream>>>(
      comp, out, mlp_q, mlp_res_nw, mlp_norm_w, xnB, rows);
  gemm_bf16_wmma<<<gemm_grid, 256, 0, stream>>>(xnB, wbfM, out, out, rows, d, d);
}
